// GC_Transformer_52398601011881
// MI455X (gfx1250) — matrix operand order repaired and checked by reading, not compiled, not run on hardware
//
#include <hip/hip_runtime.h>
#include <hip/hip_bf16.h>
#include <math.h>

// ---------------- model constants ----------------
#define Bb 32
#define Nn 1200
#define Ee 38400
#define BN (Bb*Nn)          // 38400
#define BE (Bb*Ee)          // 1228800
#define IN_DIM 17
#define GCN_OUT 14
#define HID 32
#define HEADS 2
#define FF 128
#define HEAD_DIM 16
#define QT 75               // N / 16

typedef __attribute__((ext_vector_type(16))) _Float16 v16h;
typedef __attribute__((ext_vector_type(8)))  float    v8f;
typedef __attribute__((ext_vector_type(4)))  unsigned int u32x4;
typedef __attribute__((ext_vector_type(4)))  int          i32x4;
typedef __attribute__((ext_vector_type(8)))  int          i32x8;

static inline int cdiv(int a, int b) { return (a + b - 1) / b; }

// ---------------- GCN front-end ----------------

__global__ void zero_kernel(float* p, int n) {
  int i = blockIdx.x * blockDim.x + threadIdx.x;
  if (i < n) p[i] = 0.f;
}

__global__ void build_xf_kernel(const float* __restrict__ pm25,
                                const float* __restrict__ feat,
                                float* __restrict__ xf) {
  int bn = blockIdx.x * blockDim.x + threadIdx.x;
  if (bn >= BN) return;
  int b = bn / Nn, n = bn % Nn;
  xf[bn*IN_DIM + 0] = pm25[((size_t)b*24 + 23)*Nn + n];
  const float* f = feat + (((size_t)b*72 + 71)*Nn + n)*16;
#pragma unroll
  for (int c = 0; c < 16; ++c) xf[bn*IN_DIM + 1 + c] = f[c];
}

__global__ void deg_kernel(const int* __restrict__ ei, float* __restrict__ deg) {
  int e = blockIdx.x * blockDim.x + threadIdx.x;
  if (e >= BE) return;
  atomicAdd(&deg[ei[BE + e]], 1.0f);
}

__global__ void dis_kernel(float* __restrict__ deg) {
  int i = blockIdx.x * blockDim.x + threadIdx.x;
  if (i >= BN) return;
  float d = deg[i];
  deg[i] = (d > 0.f) ? rsqrtf(fmaxf(d, 1.0f)) : 0.f;
}

__global__ void scatter_kernel(const int* __restrict__ ei,
                               const float* __restrict__ dis,
                               const float* __restrict__ xf,
                               float* __restrict__ tx1) {
  int e = blockIdx.x * blockDim.x + threadIdx.x;
  if (e >= BE) return;
  int s = ei[e], d = ei[BE + e];
  float norm = -(dis[s] * dis[d]);
  const float* xs = xf + (size_t)s * IN_DIM;
  float* td = tx1 + (size_t)d * IN_DIM;
#pragma unroll
  for (int c = 0; c < IN_DIM; ++c) atomicAdd(&td[c], norm * xs[c]);
}

// GCN sigmoid + concat + input projection + positional encoding -> h (BN,32)
__global__ void gcn_in_kernel(const float* __restrict__ xf,
                              const float* __restrict__ tx1,
                              const float* __restrict__ w0,
                              const float* __restrict__ w1,
                              const float* __restrict__ bc,
                              const float* __restrict__ w_in,
                              const float* __restrict__ b_in,
                              float* __restrict__ h) {
  int bn = blockIdx.x * blockDim.x + threadIdx.x;
  if (bn >= BN) return;
  int n = bn % Nn;
  float x31[IN_DIM + GCN_OUT];
  float xv[IN_DIM], tv[IN_DIM];
#pragma unroll
  for (int k = 0; k < IN_DIM; ++k) {
    xv[k] = xf[bn*IN_DIM + k];
    tv[k] = tx1[bn*IN_DIM + k];
    x31[k] = xv[k];
  }
#pragma unroll
  for (int j = 0; j < GCN_OUT; ++j) {
    float acc = bc[j];
#pragma unroll
    for (int k = 0; k < IN_DIM; ++k)
      acc += xv[k] * w0[k*GCN_OUT + j] + tv[k] * w1[k*GCN_OUT + j];
    x31[IN_DIM + j] = 1.0f / (1.0f + __expf(-acc));
  }
  const float ln1e4_d = 9.210340371976184f / 32.0f;
#pragma unroll 4
  for (int j = 0; j < HID; ++j) {
    float freq = __expf(-(float)(2 * (j >> 1)) * ln1e4_d);
    float ang = (float)n * freq;
    float pe = (j & 1) ? __cosf(ang) : __sinf(ang);
    float acc = b_in[j] + pe;
#pragma unroll
    for (int k = 0; k < IN_DIM + GCN_OUT; ++k) acc += x31[k] * w_in[k*HID + j];
    h[(size_t)bn*HID + j] = acc;
  }
}

// ---------------- transformer pieces ----------------

__global__ void qkv_kernel(const float* __restrict__ h,
                           const float* __restrict__ w,   // (32,96)
                           const float* __restrict__ bias,
                           float* __restrict__ qkv) {
  int tid = blockIdx.x * blockDim.x + threadIdx.x;
  if (tid >= BN * 96) return;
  int bn = tid / 96, j = tid % 96;
  const float* hr = h + (size_t)bn * HID;
  float acc = bias[j];
#pragma unroll
  for (int k = 0; k < HID; ++k) acc += hr[k] * w[k*96 + j];
  qkv[(size_t)bn*96 + j] = acc;
}

// ---- Tensor Data Mover: async 2D tile load (rows x 16 f32) into LDS ----
// D# per CDNA5 ISA: group0 = {flags, lds_addr,
// global_addr[31:0], global_addr[56:32] | type<<30}; group1 packs data_size,
// dims, strides. tensor_dim1 = rows_remaining gives zero-fill past N.
__device__ __forceinline__ void tdm_load_tile(const float* gsrc, float* ldst,
                                              int rows_remaining) {
  unsigned long long ga = (unsigned long long)(uintptr_t)gsrc;
  unsigned int lds = (unsigned int)(uintptr_t)ldst;   // LDS aperture: low 32b = offset
  u32x4 g0;
  g0[0] = 1u;                                          // count=1, user descriptor
  g0[1] = lds;
  g0[2] = (unsigned int)(ga & 0xFFFFFFFFu);
  g0[3] = (unsigned int)((ga >> 32) & 0x01FFFFFFu) | 0x80000000u;   // type=2
  i32x8 g1;
  g1[0] = (int)(2u << 16);                             // data_size = 4 bytes
  g1[1] = (int)(16u << 16);                            // tensor_dim0 = 16 (w1[31:16])
  g1[2] = (int)(((unsigned)rows_remaining & 0xFFFFu) << 16); // tensor_dim1[15:0]
  g1[3] = (int)(16u << 16);                            // tile_dim0 = 16 (w3[31:16])
  g1[4] = 32;                                          // tile_dim1 = 32, tile_dim2 = 0
  g1[5] = 96;                                          // tensor_dim0_stride = 96 elems
  g1[6] = 0;
  g1[7] = 0;
  i32x4 z4 = {0, 0, 0, 0};
#if __clang_major__ >= 23
  i32x8 z8 = {0, 0, 0, 0, 0, 0, 0, 0};
  __builtin_amdgcn_tensor_load_to_lds(g0, g1, z4, z4, z8, 0);
#else
  __builtin_amdgcn_tensor_load_to_lds(g0, g1, z4, z4, 0);
#endif
}

// Flash attention: one wave (32 lanes) per (b, head, 16-row q tile).
// WMMA f32_16x16x32_f16: QK^T with head-dim K padded 16->32, PV with exact K=32.
// K/V tiles stream via the Tensor Data Mover, double buffered (TENSORcnt).
__launch_bounds__(32)
__global__ void attn_kernel(const float* __restrict__ qkv,
                            float* __restrict__ attn_out) {
  int blk = blockIdx.x;
  int qt = blk % QT;
  int hh = (blk / QT) % HEADS;
  int b  = blk / (QT * HEADS);
  int lane = threadIdx.x;
  int n16 = lane & 15;
  int half = lane >> 4;

  __shared__ float lq[16*16];
  __shared__ float lkv[2][2][32*16];   // [buffer][0=K,1=V][row*16+dim], f32
  __shared__ _Float16 lp[16*32];

  const float* qbase = qkv + (size_t)b*Nn*96 + hh*HEAD_DIM;
  const float* kbase = qbase + 32;
  const float* vbase = qbase + 64;
  int q0 = qt * 16;

  for (int i = lane; i < 256; i += 32) {
    int r = i >> 4, c = i & 15;
    lq[i] = qbase[(size_t)(q0 + r)*96 + c];
  }
  // prologue: TDM loads for tile 0 (K then V) into buffer 0
  tdm_load_tile(kbase, &lkv[0][0][0], Nn);
  tdm_load_tile(vbase, &lkv[0][1][0], Nn);
  __syncthreads();

  // A fragment of Q (16x32 f16, head dims 16..31 zero-padded)
  v16h aq;
#pragma unroll
  for (int i = 0; i < 16; ++i) {
    int v = i >> 1, p = i & 1;
    int K = (v < 4) ? (2*v + 8*half + p) : (16 + 2*(v-4) + 8*half + p);
    aq[i] = (K < HEAD_DIM) ? (_Float16)lq[n16*16 + K] : (_Float16)0.f;
  }

  float m[8], lsum[8];
  v8f oacc = {0.f,0.f,0.f,0.f,0.f,0.f,0.f,0.f};
#pragma unroll
  for (int v = 0; v < 8; ++v) { m[v] = -1e30f; lsum[v] = 0.f; }
  const float scale = 0.25f;   // 1/sqrt(16)

  for (int kt = 0; kt < 38; ++kt) {          // ceil(1200/32)
    int kb = kt * 32;
    // issue next tile's TDM loads, then wait for the current tile to land
    if (kt + 1 < 38) {
      int nb = (kt + 1) & 1;
      tdm_load_tile(kbase + (size_t)(kb + 32)*96, &lkv[nb][0][0], Nn - (kb + 32));
      tdm_load_tile(vbase + (size_t)(kb + 32)*96, &lkv[nb][1][0], Nn - (kb + 32));
      __builtin_amdgcn_s_wait_tensorcnt(2);
    } else {
      __builtin_amdgcn_s_wait_tensorcnt(0);
    }
    __syncthreads();
    const float* lk = &lkv[kt & 1][0][0];
    const float* lv = &lkv[kt & 1][1][0];

    // S = Q K^T for two 16-column subtiles
    v8f s0, s1;
#pragma unroll
    for (int t = 0; t < 2; ++t) {
      v16h bk;
#pragma unroll
      for (int i = 0; i < 16; ++i) {
        int v = i >> 1, p = i & 1;
        int K = (v < 4) ? (2*v + 8*half + p) : (16 + 2*(v-4) + 8*half + p);   // B-matrix K index = head dim
        bk[i] = (K < HEAD_DIM) ? (_Float16)lk[(t*16 + n16)*16 + K] : (_Float16)0.f;
      }
      v8f cz = {0.f,0.f,0.f,0.f,0.f,0.f,0.f,0.f};
      v8f s = __builtin_amdgcn_wmma_f32_16x16x32_f16(
          false, aq, false, bk, (short)0, cz, false, false);
      if (t == 0) s0 = s; else s1 = s;
    }

    // online softmax update over these 32 columns (zero-filled tail masked)
    float p0[8], p1[8];
#pragma unroll
    for (int v = 0; v < 8; ++v) {
      float x0 = s0[v] * scale;
      float x1 = s1[v] * scale;
      if (kb + n16 >= Nn)      x0 = -1e30f;
      if (kb + 16 + n16 >= Nn) x1 = -1e30f;
      float mc = fmaxf(x0, x1);
      for (int off = 1; off < 16; off <<= 1) mc = fmaxf(mc, __shfl_xor(mc, off, 16));
      float mn = fmaxf(m[v], mc);
      float corr = __expf(m[v] - mn);
      float e0 = __expf(x0 - mn);
      float e1 = __expf(x1 - mn);
      float rs = e0 + e1;
      for (int off = 1; off < 16; off <<= 1) rs += __shfl_xor(rs, off, 16);
      lsum[v] = lsum[v] * corr + rs;
      oacc[v] *= corr;
      m[v] = mn;
      p0[v] = e0; p1[v] = e1;
    }

    // stage P (16x32 f16) through LDS to A-layout
#pragma unroll
    for (int v = 0; v < 8; ++v) {
      int r = v + 8*half;
      lp[r*32 + n16]      = (_Float16)p0[v];
      lp[r*32 + 16 + n16] = (_Float16)p1[v];
    }
    __syncthreads();

    v16h ap, bv;
#pragma unroll
    for (int i = 0; i < 16; ++i) {
      int v = i >> 1, p = i & 1;
      int Ka = (v < 4) ? (2*v + 8*half + p) : (16 + 2*(v-4) + 8*half + p);
      ap[i] = lp[n16*32 + Ka];
      int Kb = (v < 4) ? (2*v + 8*half + p) : (16 + 2*(v-4) + 8*half + p);   // k-row within tile, exact K=32
      bv[i] = (_Float16)lv[Kb*16 + n16];
    }
    oacc = __builtin_amdgcn_wmma_f32_16x16x32_f16(
        false, ap, false, bv, (short)0, oacc, false, false);
    __syncthreads();   // drain DS reads before this buffer is re-targeted by TDM
  }

  float* obase = attn_out + (size_t)b*Nn*HID + hh*HEAD_DIM;
#pragma unroll
  for (int v = 0; v < 8; ++v) {
    int r = q0 + v + 8*half;
    obase[(size_t)r*HID + n16] = oacc[v] / lsum[v];
  }
}

// pre = h + attn_out @ wo + bo
__global__ void addproj_kernel(const float* __restrict__ att,
                               const float* __restrict__ w,  // (32,32)
                               const float* __restrict__ bias,
                               const float* __restrict__ h,
                               float* __restrict__ pre) {
  int tid = blockIdx.x * blockDim.x + threadIdx.x;
  if (tid >= BN * HID) return;
  int bn = tid / HID, j = tid % HID;
  const float* ar = att + (size_t)bn * HID;
  float acc = bias[j] + h[(size_t)bn*HID + j];
#pragma unroll
  for (int k = 0; k < HID; ++k) acc += ar[k] * w[k*HID + j];
  pre[(size_t)bn*HID + j] = acc;
}

__global__ void ln_kernel(const float* __restrict__ x,
                          const float* __restrict__ g,
                          const float* __restrict__ bta,
                          float* __restrict__ out) {
  int bn = blockIdx.x * blockDim.x + threadIdx.x;
  if (bn >= BN) return;
  float v[HID]; float mu = 0.f;
#pragma unroll
  for (int j = 0; j < HID; ++j) { v[j] = x[(size_t)bn*HID + j]; mu += v[j]; }
  mu *= (1.0f / HID);
  float var = 0.f;
#pragma unroll
  for (int j = 0; j < HID; ++j) { float d = v[j] - mu; var += d * d; }
  var *= (1.0f / HID);
  float inv = rsqrtf(var + 1e-5f);
#pragma unroll
  for (int j = 0; j < HID; ++j)
    out[(size_t)bn*HID + j] = (v[j] - mu) * inv * g[j] + bta[j];
}

__global__ void ff1_kernel(const float* __restrict__ h,
                           const float* __restrict__ w,  // (32,128)
                           const float* __restrict__ bias,
                           float* __restrict__ ffm) {
  int tid = blockIdx.x * blockDim.x + threadIdx.x;
  if (tid >= BN * FF) return;
  int bn = tid / FF, f = tid % FF;
  const float* hr = h + (size_t)bn * HID;
  float acc = bias[f];
#pragma unroll
  for (int k = 0; k < HID; ++k) acc += hr[k] * w[k*FF + f];
  ffm[(size_t)bn*FF + f] = fmaxf(acc, 0.f);
}

__global__ void ff2_kernel(const float* __restrict__ ffm,
                           const float* __restrict__ w,  // (128,32)
                           const float* __restrict__ bias,
                           const float* __restrict__ h,
                           float* __restrict__ pre) {
  int tid = blockIdx.x * blockDim.x + threadIdx.x;
  if (tid >= BN * HID) return;
  int bn = tid / HID, j = tid % HID;
  const float* fr = ffm + (size_t)bn * FF;
  float acc = bias[j] + h[(size_t)bn*HID + j];
#pragma unroll 8
  for (int k = 0; k < FF; ++k) acc += fr[k] * w[k*HID + j];
  pre[(size_t)bn*HID + j] = acc;
}

__global__ void out_kernel(const float* __restrict__ h,
                           const float* __restrict__ w,  // (32,1)
                           const float* __restrict__ bias,
                           float* __restrict__ out) {
  int bn = blockIdx.x * blockDim.x + threadIdx.x;
  if (bn >= BN) return;
  const float* hr = h + (size_t)bn * HID;
  float acc = bias[0];
#pragma unroll
  for (int j = 0; j < HID; ++j) acc += hr[j] * w[j];
  out[bn] = acc;
}

// ---------------- launcher ----------------

extern "C" void kernel_launch(void* const* d_in, const int* in_sizes, int n_in,
                              void* d_out, int out_size, void* d_ws, size_t ws_size,
                              hipStream_t stream) {
  const float* pm25   = (const float*)d_in[0];
  const float* feat   = (const float*)d_in[1];
  const int*   ei     = (const int*)  d_in[2];
  const float* wch0   = (const float*)d_in[3];
  const float* wch1   = (const float*)d_in[4];
  const float* bch    = (const float*)d_in[5];
  const float* w_in   = (const float*)d_in[6];
  const float* b_in   = (const float*)d_in[7];
  const float* wqkv   = (const float*)d_in[8];
  const float* bqkv   = (const float*)d_in[9];
  const float* wo     = (const float*)d_in[10];
  const float* bo     = (const float*)d_in[11];
  const float* ln1g   = (const float*)d_in[12];
  const float* ln1b   = (const float*)d_in[13];
  const float* wff1   = (const float*)d_in[14];
  const float* bff1   = (const float*)d_in[15];
  const float* wff2   = (const float*)d_in[16];
  const float* bff2   = (const float*)d_in[17];
  const float* ln2g   = (const float*)d_in[18];
  const float* ln2b   = (const float*)d_in[19];
  const float* w_out  = (const float*)d_in[20];
  const float* b_out  = (const float*)d_in[21];
  float* out = (float*)d_out;

  // workspace layout (floats)
  float* ws  = (float*)d_ws;
  float* xf  = ws;                        // BN*17
  float* deg = xf  + (size_t)BN*IN_DIM;   // BN
  float* tx1 = deg + (size_t)BN;          // BN*17
  float* h   = tx1 + (size_t)BN*IN_DIM;   // BN*32
  float* qkv = h   + (size_t)BN*HID;      // BN*96
  float* att = qkv + (size_t)BN*96;       // BN*32
  float* pre = att + (size_t)BN*HID;      // BN*32
  float* ffm = pre + (size_t)BN*HID;      // BN*128

  const int TB = 256;

  // zero atomic accumulation buffers every call (graph-replay safe)
  zero_kernel<<<cdiv(BN*(IN_DIM+1), TB), TB, 0, stream>>>(deg, BN*(IN_DIM+1));

  build_xf_kernel<<<cdiv(BN, TB), TB, 0, stream>>>(pm25, feat, xf);
  deg_kernel<<<cdiv(BE, TB), TB, 0, stream>>>(ei, deg);
  dis_kernel<<<cdiv(BN, TB), TB, 0, stream>>>(deg);
  scatter_kernel<<<cdiv(BE, TB), TB, 0, stream>>>(ei, deg, xf, tx1);
  gcn_in_kernel<<<cdiv(BN, TB), TB, 0, stream>>>(xf, tx1, wch0, wch1, bch,
                                                 w_in, b_in, h);

  for (int l = 0; l < 2; ++l) {
    qkv_kernel<<<cdiv(BN*96, TB), TB, 0, stream>>>(
        h, wqkv + (size_t)l*HID*96, bqkv + (size_t)l*96, qkv);
    attn_kernel<<<Bb*HEADS*QT, 32, 0, stream>>>(qkv, att);
    addproj_kernel<<<cdiv(BN*HID, TB), TB, 0, stream>>>(
        att, wo + (size_t)l*HID*HID, bo + (size_t)l*HID, h, pre);
    ln_kernel<<<cdiv(BN, TB), TB, 0, stream>>>(
        pre, ln1g + (size_t)l*HID, ln1b + (size_t)l*HID, h);
    ff1_kernel<<<cdiv(BN*FF, TB), TB, 0, stream>>>(
        h, wff1 + (size_t)l*HID*FF, bff1 + (size_t)l*FF, ffm);
    ff2_kernel<<<cdiv(BN*HID, TB), TB, 0, stream>>>(
        ffm, wff2 + (size_t)l*FF*HID, bff2 + (size_t)l*HID, h, pre);
    ln_kernel<<<cdiv(BN, TB), TB, 0, stream>>>(
        pre, ln2g + (size_t)l*HID, ln2b + (size_t)l*HID, h);
  }

  out_kernel<<<cdiv(BN, TB), TB, 0, stream>>>(h, w_out, b_out, out);
}
